// Topkblock_55989193671529
// MI455X (gfx1250) — matrix operand order repaired and checked by reading, not compiled, not run on hardware
//
#include <hip/hip_runtime.h>

// CDNA5 / gfx1250: wave32, WMMA bf16 16x16x32, f32 accumulate.
typedef __attribute__((ext_vector_type(16))) __bf16 v16bf;
typedef __attribute__((ext_vector_type(8)))  float  v8f;

#define CIN    128
#define COUT   256
#define HWSZ   4096             // H*W
#define KTOT   1152             // 9 taps * 128 Cin  (K reordered tap-major)
#define KSTEP  32
#define NCHUNK (KTOT / KSTEP)   // 36
#define NPIX   128              // pixels (N columns) per workgroup = two W rows
#define NTILE  (NPIX / 16)      // 8 N-tiles per wave
#define PPW    (NPIX / 8)       // 16 pixels per wave for top-k

__device__ __forceinline__ unsigned short f2bf(float f) {
  unsigned u = __float_as_uint(f);
  unsigned r = u + 0x7FFFu + ((u >> 16) & 1u);   // round-to-nearest-even
  return (unsigned short)(r >> 16);
}

// map float -> uint with same total order (for radix select / compare)
__device__ __forceinline__ unsigned ordmap(float f) {
  unsigned u = __float_as_uint(f);
  return (u & 0x80000000u) ? ~u : (u | 0x80000000u);
}

struct __align__(16) Smem {
  union {
    struct {                             // double-buffered GEMM staging
      unsigned short A[2][COUT * KSTEP]; // 2 x (256 x 32) bf16   (32 KB)
      unsigned short B[2][NPIX * KSTEP]; // 2 x (128 px x 32 k)   (16 KB)
    } g;
    float O[NPIX][COUT + 4];             // post-noise activations (133 KB)
  } u;
  float    absb[COUT];
  float    rbias[COUT];
  unsigned threshu[NPIX];
};

union FragU { uint4 q[2]; v16bf v; };

// ---------------------------------------------------------------------------
// Prep: weight f32 (Cout,Cin,3,3) -> bf16 A-matrix, K reordered tap-major:
//   wbf[cout][tap*128 + ci] = w[cout][ci][tap]   (tap = r*3+s)
// so each 32-wide K-chunk of the GEMM has a single (r,s) tap.
// Also absbias[cout] = sum|w|.
// ---------------------------------------------------------------------------
__global__ __launch_bounds__(128)
void prep_weights(const float* __restrict__ wsrc,
                  unsigned short* __restrict__ wbf,
                  float* __restrict__ absb) {
  const int cout = blockIdx.x;
  const int ci = threadIdx.x;            // 0..127
  float s = 0.f;
  #pragma unroll
  for (int tap = 0; tap < 9; ++tap) {
    float v = wsrc[cout * KTOT + ci * 9 + tap];
    s += fabsf(v);
    wbf[cout * KTOT + tap * 128 + ci] = f2bf(v);
  }
  __shared__ float red[128];
  red[ci] = s;
  __syncthreads();
  for (int st = 64; st > 0; st >>= 1) {
    if (ci < st) red[ci] += red[ci + st];
    __syncthreads();
  }
  if (ci == 0) absb[cout] = red[0];
}

// ---------------------------------------------------------------------------
// Fused implicit-GEMM conv3x3 + noise + per-pixel channel top-64 + bias/ReLU.
// One workgroup (8 waves) owns all 256 Cout for 128 pixels (two W rows).
// Double-buffered software pipeline: stage chunk i+1 (async A + B gather)
// overlapped with the WMMA block of chunk i; one barrier per chunk.
// ---------------------------------------------------------------------------
__global__ __launch_bounds__(256)
void conv_topk_fused(const float* __restrict__ x,
                     const unsigned short* __restrict__ wbf,
                     const float* __restrict__ absb_g,
                     const float* __restrict__ relu_bias,
                     const float* __restrict__ noise_u,
                     float* __restrict__ out) {
  __shared__ Smem sm;
  const int t    = threadIdx.x;
  const int lane = t & 31;
  const int wv   = t >> 5;               // wave 0..7
  const int wg   = blockIdx.x;           // 0..1023 pixel blocks
  const int hilo = (lane < 16) ? 0 : 1;  // lane half (WMMA layout)
  const int l16  = lane & 15;

  sm.absb[t]  = absb_g[t];
  sm.rbias[t] = relu_bias[t];

  v8f acc[2][NTILE];
  {
    v8f z = {0.f, 0.f, 0.f, 0.f, 0.f, 0.f, 0.f, 0.f};
    #pragma unroll
    for (int mi = 0; mi < 2; ++mi)
      #pragma unroll
      for (int ni = 0; ni < NTILE; ++ni) acc[mi][ni] = z;
  }

  // im2col staging role: pixel nl (0..127), k-group kgrp (16 ci's each)
  const int nl   = t & (NPIX - 1);
  const int kgrp = t >> 7;               // 0..1
  const int ng   = wg * NPIX + nl;
  const int bb   = ng >> 12;             // batch  (4096 pixels / image)
  const int hw   = ng & (HWSZ - 1);
  const int hh0  = (hw >> 6) - 1;
  const int ww0  = (hw & 63) - 1;
  const int xrow = (bb * CIN) << 12;     // 32-bit offsets throughout

  // LDS byte offsets of this thread's A-staging rows (generic ptr low 32
  // bits are the LDS offset per the flat-aperture rule).
  const unsigned ldsA0 = (unsigned)(unsigned long long)(void*)&sm.u.g.A[0][t * KSTEP];
  const unsigned ldsA1 = (unsigned)(unsigned long long)(void*)&sm.u.g.A[1][t * KSTEP];
  const unsigned long long gaBase = (unsigned long long)(wbf + t * KTOT);

  // stage chunk c into buffer c&1 (async A copy + im2col B gather)
  auto stage = [&](int c) {
    const unsigned p = (unsigned)(c & 1);
    // A: 64B of bf16 weights, memory -> LDS with no VGPR round-trip.
    {
      const unsigned lA = p ? ldsA1 : ldsA0;
      const unsigned long long ga = gaBase + (unsigned)(c * (KSTEP * 2));
      asm volatile(
          "global_load_async_to_lds_b128 %0, %1, off\n\t"
          "global_load_async_to_lds_b128 %0, %1, off offset:16\n\t"
          "global_load_async_to_lds_b128 %0, %1, off offset:32\n\t"
          "global_load_async_to_lds_b128 %0, %1, off offset:48"
          :: "v"(lA), "v"(ga) : "memory");
    }
    // B: tap uniform across the chunk; single guarded 16-load clause.
    {
      const int tap = c >> 2;            // 0..8
      const int r   = tap / 3;
      const int s   = tap - r * 3;
      const int hh  = hh0 + r;
      const int ww  = ww0 + s;
      const int ci  = ((c & 3) << 5) + (kgrp << 4);
      const float* px = x + (xrow + (ci << 12) + (hh << 6) + ww);
      float va[16];
      #pragma unroll
      for (int j = 0; j < 16; ++j) va[j] = 0.f;
      if (((unsigned)hh < 64u) && ((unsigned)ww < 64u)) {
        #pragma unroll
        for (int j = 0; j < 16; ++j) va[j] = px[j << 12];
      }
      uint4 pv0, pv1;
      pv0.x = (unsigned)f2bf(va[0])  | ((unsigned)f2bf(va[1])  << 16);
      pv0.y = (unsigned)f2bf(va[2])  | ((unsigned)f2bf(va[3])  << 16);
      pv0.z = (unsigned)f2bf(va[4])  | ((unsigned)f2bf(va[5])  << 16);
      pv0.w = (unsigned)f2bf(va[6])  | ((unsigned)f2bf(va[7])  << 16);
      pv1.x = (unsigned)f2bf(va[8])  | ((unsigned)f2bf(va[9])  << 16);
      pv1.y = (unsigned)f2bf(va[10]) | ((unsigned)f2bf(va[11]) << 16);
      pv1.z = (unsigned)f2bf(va[12]) | ((unsigned)f2bf(va[13]) << 16);
      pv1.w = (unsigned)f2bf(va[14]) | ((unsigned)f2bf(va[15]) << 16);
      unsigned short* brow = &sm.u.g.B[p][nl * KSTEP + kgrp * 16];
      *(uint4*)brow       = pv0;
      *(uint4*)(brow + 8) = pv1;
    }
  };

  stage(0);                              // prologue

  for (int c = 0; c < NCHUNK; ++c) {
    // all async A-copies issued so far are complete before we publish
    asm volatile("s_wait_asynccnt 0x0" ::: "memory");
    __syncthreads();
    if (c + 1 < NCHUNK) stage(c + 1);    // overlaps with WMMAs below

    const unsigned short* Ab = &sm.u.g.A[c & 1][0];
    const unsigned short* Bb = &sm.u.g.B[c & 1][0];

    // A fragments resident (16 VGPRs); B tiles streamed one at a time to
    // bound register pressure at acc(128) + fa(16) + fb(8).
    const int kbA = hilo ? 16 : 0;       // A: K {kb..kb+15}
    FragU fa[2];
    #pragma unroll
    for (int mi = 0; mi < 2; ++mi) {
      const int m = wv * 32 + mi * 16 + l16;
      fa[mi].q[0] = *(const uint4*)(Ab + m * KSTEP + kbA);
      fa[mi].q[1] = *(const uint4*)(Ab + m * KSTEP + kbA + 8);
    }
    const int kbB = hilo ? 16 : 0;       // B: lanes 0-15 K0..15, 16-31 K16..31
    #pragma unroll
    for (int ni = 0; ni < NTILE; ++ni) {
      FragU fb;
      const uint4* pB = (const uint4*)(Bb + (ni * 16 + l16) * KSTEP + kbB);
      fb.q[0] = pB[0]; fb.q[1] = pB[1];
      #pragma unroll
      for (int mi = 0; mi < 2; ++mi) {
        acc[mi][ni] = __builtin_amdgcn_wmma_f32_16x16x32_bf16(
            false, fa[mi].v, false, fb.v, (short)0, acc[mi][ni], false, false);
      }
    }
  }

  __syncthreads();   // A/B region dead; union region becomes O

  // --- add structured noise, spill to LDS O[pixel][cout] (2x b128/tile) ---
  const float NS = 8.0f / 255.0f;        // NOISE_SCALE, GAMMA = 1
  #pragma unroll
  for (int mi = 0; mi < 2; ++mi) {
    const int cb = wv * 32 + mi * 16 + hilo * 8;   // 8 consecutive couts
    float ab[8];
    *(float4*)&ab[0] = *(const float4*)&sm.absb[cb];
    *(float4*)&ab[4] = *(const float4*)&sm.absb[cb + 4];
    #pragma unroll
    for (int ni = 0; ni < NTILE; ++ni) {
      const int n_local = ni * 16 + l16;
      const int png = wg * NPIX + n_local;
      const int pb  = png >> 12;
      const int phw = png & (HWSZ - 1);
      const float* nup = noise_u + ((((size_t)pb * COUT + cb) << 12) + phw);
      float vv[8];
      #pragma unroll
      for (int i = 0; i < 8; ++i) {      // noise: base + i*16KiB (imm offsets)
        float nu = nup[(size_t)i << 12];
        float v  = acc[mi][ni][i] + ab[i] * NS * (2.0f * nu - 1.0f);
        acc[mi][ni][i] = v;
        vv[i] = v;
      }
      float* orow = &sm.u.O[n_local][cb];
      *(float4*)orow       = *(const float4*)&vv[0];
      *(float4*)(orow + 4) = *(const float4*)&vv[4];
    }
  }
  __syncthreads();

  // --- exact 64th-largest per pixel via 32-bit MSB-first radix select ---
  // wave wv owns pixels wv*PPW .. wv*PPW+PPW-1 ; each lane holds 8 channels.
  for (int pp = 0; pp < PPW; ++pp) {
    const int pix = wv * PPW + pp;
    const float* row = &sm.u.O[pix][0];
    float4 v0 = *(const float4*)(&row[lane * 8]);
    float4 v1 = *(const float4*)(&row[lane * 8 + 4]);
    unsigned uv[8];
    uv[0] = ordmap(v0.x); uv[1] = ordmap(v0.y); uv[2] = ordmap(v0.z); uv[3] = ordmap(v0.w);
    uv[4] = ordmap(v1.x); uv[5] = ordmap(v1.y); uv[6] = ordmap(v1.z); uv[7] = ordmap(v1.w);
    unsigned prefix = 0u;
    int remaining = 64;                  // K
    for (int bit = 31; bit >= 0; --bit) {
      const unsigned desired = prefix | (1u << bit);
      const unsigned mask = ~((1u << bit) - 1u);
      int cnt = 0;
      #pragma unroll
      for (int j = 0; j < 8; ++j) cnt += ((uv[j] & mask) == desired) ? 1 : 0;
      #pragma unroll
      for (int off = 16; off > 0; off >>= 1) cnt += __shfl_xor(cnt, off, 32);
      if (cnt >= remaining) prefix = desired;
      else                  remaining -= cnt;
    }
    if (lane == 0) sm.threshu[pix] = prefix;   // ordmap of K-th largest value
  }
  __syncthreads();

  // --- threshold (>= kth value, same tie semantics as reference),
  //     + relu_bias, ReLU, coalesced stores from live accumulators ---
  #pragma unroll
  for (int mi = 0; mi < 2; ++mi) {
    const int cb = wv * 32 + mi * 16 + hilo * 8;
    float rb[8];
    *(float4*)&rb[0] = *(const float4*)&sm.rbias[cb];
    *(float4*)&rb[4] = *(const float4*)&sm.rbias[cb + 4];
    #pragma unroll
    for (int ni = 0; ni < NTILE; ++ni) {
      const int n_local = ni * 16 + l16;
      const unsigned th = sm.threshu[n_local];
      const int png = wg * NPIX + n_local;
      const int pb  = png >> 12;
      const int phw = png & (HWSZ - 1);
      float* op = out + ((((size_t)pb * COUT + cb) << 12) + phw);
      #pragma unroll
      for (int i = 0; i < 8; ++i) {
        const float v = acc[mi][ni][i];
        const float kept = (ordmap(v) >= th) ? v : 0.0f;
        op[(size_t)i << 12] = fmaxf(kept + rb[i], 0.0f);
      }
    }
  }
}

// ---------------------------------------------------------------------------
extern "C" void kernel_launch(void* const* d_in, const int* in_sizes, int n_in,
                              void* d_out, int out_size, void* d_ws, size_t ws_size,
                              hipStream_t stream) {
  const float* x         = (const float*)d_in[0];   // (32,128,64,64)
  const float* weight    = (const float*)d_in[1];   // (256,128,3,3)
  const float* relu_bias = (const float*)d_in[2];   // (256,)
  const float* noise_u   = (const float*)d_in[3];   // (32,256,64,64)
  float* out = (float*)d_out;                       // (32,256,64,64)

  unsigned short* wbf  = (unsigned short*)d_ws;                       // 576 KB
  float*          absb = (float*)((char*)d_ws + (size_t)COUT * KTOT * 2);

  prep_weights<<<COUT, 128, 0, stream>>>(weight, wbf, absb);

  const int nblocks = (32 * HWSZ) / NPIX;           // 131072/128 = 1024
  conv_topk_fused<<<nblocks, 256, 0, stream>>>(x, wbf, absb, relu_bias,
                                               noise_u, out);
}
